// LSTMAttn_78812649882277
// MI455X (gfx1250) — compile-verified, matrix operand order checked by reading, not run on hardware
//
#include <hip/hip_runtime.h>

// ---------------------------------------------------------------------------
// LSTM(10 layers, B=256,S=512,IN=27,H=128) + additive attention + FC  -> [B,7]
//   * All GEMMs via v_wmma_f32_16x16x32_f16 (CDNA5 wave32 WMMA).
//   * Per layer: (1) parallel WMMA GEMM xg = in @ w_ih^T + bias (f16, L2-res)
//                (2) serial scan: 16 WGs (16 batch rows each) hold w_hh
//                    B-fragments in VGPRs, h double-buffered in LDS, c in
//                    registers; gate-interleaved N-tiles keep i/f/g/o in-lane.
//                    xg[t+1] register-prefetched during step t's WMMAs so the
//                    serial chain is LDS->WMMA->VALU->LDS only.
//   * Weights repacked once into the ISA 7.12.2 WMMA fragment layout.
// Workspace: ~196 MiB (xg16 128Mi + 2x hidden 32Mi + packed weights/bias).
// ---------------------------------------------------------------------------

typedef __attribute__((ext_vector_type(16))) _Float16 v16h;
typedef __attribute__((ext_vector_type(8)))  float    v8f;
typedef __attribute__((ext_vector_type(4)))  unsigned int uint4v;

union F16x16 { uint4v q[2]; v16h v; };

#define BATCH 256
#define SEQ   512
#define HID   128
#define GATES 512   /* 4*HID */
#define NIN   27
#define NLAYER 10
#define NOUT  7

// ---------------------------------------------------------------------------
// Repack f32 [N][128] row-major weights (out = in @ W^T) into the WMMA
// B-fragment layout: tile (kt,nt) -> 512 halfs; pos = lane*16 + e;
//   n = nt*16 + lane%16 ; k = kt*32 + (lane/16)*8 + (e<8 ? e : e+8)
// ---------------------------------------------------------------------------
__global__ void __launch_bounds__(256)
repack_w16(const float* __restrict__ src, _Float16* __restrict__ dst,
           int NT, int total) {
  int tid = blockIdx.x * 256 + threadIdx.x;
  if (tid >= total) return;
  int eper  = NT * 2048;             // elements per layer (== NT*16*128)
  int layer = tid / eper;
  int rem   = tid % eper;
  int tile  = rem / 512;
  int pos   = rem % 512;
  int kt = tile / NT, nt = tile % NT;
  int lane = pos / 16, e = pos % 16;
  int n = nt * 16 + (lane & 15);
  int k = kt * 32 + (lane >> 4) * 8 + (e < 8 ? e : e + 8);
  dst[tid] = (_Float16)src[(size_t)layer * (NT * 2048) + n * 128 + k];
}

__global__ void __launch_bounds__(256)
bias_sum(const float* __restrict__ bih, const float* __restrict__ bhh,
         float* __restrict__ bsum, int total) {
  int tid = blockIdx.x * 256 + threadIdx.x;
  if (tid < total) bsum[tid] = bih[tid] + bhh[tid];
}

// ---------------------------------------------------------------------------
// Layer-0 input GEMM (K = 27, tiny): xg16[row][n] = x[row]·w0[n] + bsum0[n]
// ---------------------------------------------------------------------------
__global__ void __launch_bounds__(256)
xg_layer0(const float* __restrict__ x, const float* __restrict__ w0,
          const float* __restrict__ bsum0, _Float16* __restrict__ xg) {
  int tid = blockIdx.x * 256 + threadIdx.x;   // over B*S*512
  int row = tid >> 9;
  int n   = tid & 511;
  const float* xr = x + (size_t)row * NIN;
  const float* wr = w0 + (size_t)n * NIN;
  float s = bsum0[n];
#pragma unroll
  for (int d = 0; d < NIN; ++d) s += xr[d] * wr[d];
  xg[(size_t)tid] = (_Float16)s;
}

// ---------------------------------------------------------------------------
// WMMA GEMM for layers 1..9: xg16[row][n] = hin[row]·w_ih[n] + bsum[n]
// block = one M-tile (16 rows of B*S); wave w covers N-tiles 4w..4w+3.
// ---------------------------------------------------------------------------
__global__ void __launch_bounds__(256)
gemm_xg(const _Float16* __restrict__ hin, const _Float16* __restrict__ wp,
        const float* __restrict__ bs, _Float16* __restrict__ xg) {
  int tid = threadIdx.x, w = tid >> 5, l = tid & 31;
  int m = l & 15, hi = l >> 4;
  int rowbase = blockIdx.x * 16;

  F16x16 a[4];
#pragma unroll
  for (int kt = 0; kt < 4; ++kt) {
    const uint4v* p = (const uint4v*)(hin + (size_t)(rowbase + m) * HID +
                                      kt * 32 + hi * 8);
    a[kt].q[0] = p[0];          // k = base+0..7
    a[kt].q[1] = p[2];          // k = base+16..23
  }

  v8f acc[4];
#pragma unroll
  for (int gi = 0; gi < 4; ++gi) {
    float b = bs[(w * 4 + gi) * 16 + m];
#pragma unroll
    for (int r = 0; r < 8; ++r) acc[gi][r] = b;
  }

#pragma unroll
  for (int kt = 0; kt < 4; ++kt) {
#pragma unroll
    for (int gi = 0; gi < 4; ++gi) {
      F16x16 bf;
      const uint4v* bp =
          (const uint4v*)(wp + ((kt * 32 + (w * 4 + gi)) * 512) + l * 16);
      bf.q[0] = bp[0];
      bf.q[1] = bp[1];
      acc[gi] = __builtin_amdgcn_wmma_f32_16x16x32_f16(
          false, a[kt].v, false, bf.v, (short)0, acc[gi], false, false);
    }
  }

#pragma unroll
  for (int gi = 0; gi < 4; ++gi) {
    int n = (w * 4 + gi) * 16 + m;
#pragma unroll
    for (int r = 0; r < 8; ++r) {
      int row = rowbase + r + 8 * hi;
      xg[(size_t)row * GATES + n] = (_Float16)acc[gi][r];
    }
  }
}

// ---------------------------------------------------------------------------
// Recurrent scan: one block = 16 batch rows, full 512-step recurrence.
// Wave w owns N-tiles {w, w+8, w+16, w+24} == gates i/f/g/o for hidden cols
// [16w, 16w+16) -> c/h update entirely in registers.  w_hh fragments live in
// VGPRs; h crosses waves through a double-buffered padded LDS tile (one
// barrier per step); xg[t+1] is register-prefetched during step t.
// ---------------------------------------------------------------------------
#define LDSS 136   /* padded row stride in halfs: conflict-free b128 reads */

__global__ void __launch_bounds__(256)
lstm_scan(const _Float16* __restrict__ xg, const _Float16* __restrict__ wp,
          _Float16* __restrict__ hout) {
  __shared__ _Float16 h_lds[2][16 * LDSS];
  int tid = threadIdx.x, w = tid >> 5, l = tid & 31;
  int m = l & 15, hi = l >> 4;
  int b0 = blockIdx.x * 16;
  int ncol = w * 16 + m;

  // preload the wave's 16 w_hh B-fragments into VGPRs (held across all steps)
  F16x16 bf[16];
#pragma unroll
  for (int kt = 0; kt < 4; ++kt) {
#pragma unroll
    for (int gi = 0; gi < 4; ++gi) {
      const uint4v* bp =
          (const uint4v*)(wp + ((kt * 32 + (w + gi * 8)) * 512) + l * 16);
      bf[kt * 4 + gi].q[0] = bp[0];
      bf[kt * 4 + gi].q[1] = bp[1];
    }
  }

  float c8[8];
#pragma unroll
  for (int r = 0; r < 8; ++r) c8[r] = 0.f;
  for (int i = tid; i < 16 * LDSS; i += 256) h_lds[0][i] = (_Float16)0.f;

  size_t rb[8];
#pragma unroll
  for (int r = 0; r < 8; ++r)
    rb[r] = ((size_t)(b0 + r + 8 * hi) * SEQ) * GATES + m;

  // prologue: prefetch xg for t = 0
  _Float16 xh[4][8];
#pragma unroll
  for (int gi = 0; gi < 4; ++gi) {
    int nb = (w + gi * 8) * 16;
#pragma unroll
    for (int r = 0; r < 8; ++r) xh[gi][r] = xg[rb[r] + nb];
  }

  for (int t = 0; t < SEQ; ++t) {
    const _Float16* hread = h_lds[t & 1];
    _Float16* hwrite = h_lds[(t + 1) & 1];
    __syncthreads();                 // step t-1 writes to hread visible;
                                     // also orders t-1 reads vs t writes
    F16x16 a[4];
#pragma unroll
    for (int kt = 0; kt < 4; ++kt) {
      const uint4v* p = (const uint4v*)(hread + m * LDSS + kt * 32 + hi * 8);
      a[kt].q[0] = p[0];
      a[kt].q[1] = p[2];
    }

    // consume prefetched xg into accumulators
    v8f acc[4];
#pragma unroll
    for (int gi = 0; gi < 4; ++gi)
#pragma unroll
      for (int r = 0; r < 8; ++r) acc[gi][r] = (float)xh[gi][r];

    // prefetch xg for t+1 (independent of h; overlaps the WMMAs below)
    int tn = (t + 1 < SEQ) ? t + 1 : t;
#pragma unroll
    for (int gi = 0; gi < 4; ++gi) {
      int nb = (w + gi * 8) * 16;
#pragma unroll
      for (int r = 0; r < 8; ++r)
        xh[gi][r] = xg[rb[r] + (size_t)tn * GATES + nb];
    }

#pragma unroll
    for (int kt = 0; kt < 4; ++kt) {
#pragma unroll
      for (int gi = 0; gi < 4; ++gi)
        acc[gi] = __builtin_amdgcn_wmma_f32_16x16x32_f16(
            false, a[kt].v, false, bf[kt * 4 + gi].v, (short)0, acc[gi],
            false, false);
    }

#pragma unroll
    for (int r = 0; r < 8; ++r) {
      float ig = 1.f / (1.f + __expf(-acc[0][r]));
      float fg = 1.f / (1.f + __expf(-acc[1][r]));
      float gg = tanhf(acc[2][r]);
      float og = 1.f / (1.f + __expf(-acc[3][r]));
      c8[r] = fg * c8[r] + ig * gg;
      float hv = og * tanhf(c8[r]);
      hwrite[(r + 8 * hi) * LDSS + ncol] = (_Float16)hv;
      hout[((size_t)(b0 + r + 8 * hi) * SEQ + t) * HID + ncol] = (_Float16)hv;
    }
  }
}

// ---------------------------------------------------------------------------
// Attention logits: logit[row] = v_b + sum_n v_w[n]*tanh(h[row]·attn_w[n]+ab[n])
// WMMA over N=128 (8 N-tiles, one per wave), fused tanh + v_w reduction.
// ---------------------------------------------------------------------------
__global__ void __launch_bounds__(256)
attn_logits(const _Float16* __restrict__ hseq, const _Float16* __restrict__ wp,
            const float* __restrict__ ab, const float* __restrict__ vw,
            const float* __restrict__ vb, float* __restrict__ logits) {
  __shared__ float lsum[16];
  int tid = threadIdx.x, w = tid >> 5, l = tid & 31;
  int m = l & 15, hi = l >> 4;
  int rowbase = blockIdx.x * 16;
  if (tid < 16) lsum[tid] = vb[0];
  __syncthreads();

  F16x16 a[4];
#pragma unroll
  for (int kt = 0; kt < 4; ++kt) {
    const uint4v* p = (const uint4v*)(hseq + (size_t)(rowbase + m) * HID +
                                      kt * 32 + hi * 8);
    a[kt].q[0] = p[0];
    a[kt].q[1] = p[2];
  }
  int n = w * 16 + m;
  float b = ab[n];
  v8f acc;
#pragma unroll
  for (int r = 0; r < 8; ++r) acc[r] = b;
#pragma unroll
  for (int kt = 0; kt < 4; ++kt) {
    F16x16 bfrag;
    const uint4v* bp = (const uint4v*)(wp + ((kt * 8 + w) * 512) + l * 16);
    bfrag.q[0] = bp[0];
    bfrag.q[1] = bp[1];
    acc = __builtin_amdgcn_wmma_f32_16x16x32_f16(false, a[kt].v, false,
                                                 bfrag.v, (short)0, acc,
                                                 false, false);
  }
  float vwn = vw[n];
#pragma unroll
  for (int r = 0; r < 8; ++r)
    atomicAdd(&lsum[r + 8 * hi], tanhf(acc[r]) * vwn);
  __syncthreads();
  if (tid < 16) logits[rowbase + tid] = lsum[tid];
}

// ---------------------------------------------------------------------------
// Softmax over time + weighted context + final FC.  One block per batch row.
// ---------------------------------------------------------------------------
__global__ void __launch_bounds__(256)
softmax_ctx_fc(const _Float16* __restrict__ hseq,
               const float* __restrict__ logits,
               const float* __restrict__ fcw, const float* __restrict__ fcb,
               float* __restrict__ out) {
  __shared__ float red[256];
  __shared__ float wbuf[SEQ];
  __shared__ float ctx[HID];
  int b = blockIdx.x, tid = threadIdx.x;
  float l0 = logits[(size_t)b * SEQ + tid];
  float l1 = logits[(size_t)b * SEQ + 256 + tid];
  red[tid] = fmaxf(l0, l1);
  __syncthreads();
  for (int s = 128; s > 0; s >>= 1) {
    if (tid < s) red[tid] = fmaxf(red[tid], red[tid + s]);
    __syncthreads();
  }
  float mx = red[0];
  __syncthreads();
  float e0 = __expf(l0 - mx), e1 = __expf(l1 - mx);
  wbuf[tid] = e0;
  wbuf[tid + 256] = e1;
  red[tid] = e0 + e1;
  __syncthreads();
  for (int s = 128; s > 0; s >>= 1) {
    if (tid < s) red[tid] += red[tid + s];
    __syncthreads();
  }
  float inv = 1.f / red[0];
  __syncthreads();
  if (tid < HID) {
    float acc = 0.f;
    const _Float16* hp = hseq + (size_t)b * SEQ * HID + tid;
    for (int t = 0; t < SEQ; ++t) acc += wbuf[t] * (float)hp[(size_t)t * HID];
    ctx[tid] = acc * inv;
  }
  __syncthreads();
  if (tid < NOUT) {
    float s = fcb[tid];
    const float* wr = fcw + tid * HID;
#pragma unroll 4
    for (int j = 0; j < HID; ++j) s += ctx[j] * wr[j];
    out[(size_t)b * NOUT + tid] = s;
  }
}

// ---------------------------------------------------------------------------
extern "C" void kernel_launch(void* const* d_in, const int* in_sizes, int n_in,
                              void* d_out, int out_size, void* d_ws,
                              size_t ws_size, hipStream_t stream) {
  const float* x      = (const float*)d_in[0];   // [B,S,27]
  const float* w_ih0  = (const float*)d_in[1];   // [512,27]
  const float* w_ih   = (const float*)d_in[2];   // [9,512,128]
  const float* w_hh   = (const float*)d_in[3];   // [10,512,128]
  const float* b_ih   = (const float*)d_in[4];   // [10,512]
  const float* b_hh   = (const float*)d_in[5];   // [10,512]
  const float* attn_w = (const float*)d_in[6];   // [128,128]
  const float* attn_b = (const float*)d_in[7];   // [128]
  const float* v_w    = (const float*)d_in[8];   // [1,128]
  const float* v_b    = (const float*)d_in[9];   // [1]
  const float* fc_w   = (const float*)d_in[10];  // [7,128]
  const float* fc_b   = (const float*)d_in[11];  // [7]
  float* out = (float*)d_out;

  char* p = (char*)d_ws;
  auto alloc = [&](size_t bytes) -> void* {
    void* r = (void*)p;
    p += (bytes + 255) & ~(size_t)255;
    return r;
  };
  const size_t MROWS = (size_t)BATCH * SEQ;          // 131072
  _Float16* xg16   = (_Float16*)alloc(MROWS * GATES * 2);      // 128 MiB
  _Float16* hA     = (_Float16*)alloc(MROWS * HID * 2);        // 32 MiB
  _Float16* hB     = (_Float16*)alloc(MROWS * HID * 2);        // 32 MiB
  _Float16* wp_hh  = (_Float16*)alloc((size_t)NLAYER * 65536 * 2);
  _Float16* wp_ih  = (_Float16*)alloc((size_t)(NLAYER - 1) * 65536 * 2);
  _Float16* wp_at  = (_Float16*)alloc((size_t)16384 * 2);
  float*    bsum   = (float*)alloc((size_t)NLAYER * GATES * 4);
  float*    logits = (float*)alloc(MROWS * 4);
  (void)ws_size; (void)in_sizes; (void)n_in; (void)out_size;

  // --- weight repacking into WMMA fragment layout -----------------------
  {
    int tot = NLAYER * 65536;
    repack_w16<<<(tot + 255) / 256, 256, 0, stream>>>(w_hh, wp_hh, 32, tot);
  }
  {
    int tot = (NLAYER - 1) * 65536;
    repack_w16<<<(tot + 255) / 256, 256, 0, stream>>>(w_ih, wp_ih, 32, tot);
  }
  {
    int tot = 8 * 2048;
    repack_w16<<<(tot + 255) / 256, 256, 0, stream>>>(attn_w, wp_at, 8, tot);
  }
  bias_sum<<<(NLAYER * GATES + 255) / 256, 256, 0, stream>>>(
      b_ih, b_hh, bsum, NLAYER * GATES);

  // --- layer 0 ----------------------------------------------------------
  {
    size_t tot = MROWS * GATES;
    xg_layer0<<<(unsigned)(tot / 256), 256, 0, stream>>>(x, w_ih0, bsum, xg16);
    lstm_scan<<<BATCH / 16, 256, 0, stream>>>(xg16, wp_hh, hA);
  }
  // --- layers 1..9 ------------------------------------------------------
  _Float16* cur = hA;
  _Float16* nxt = hB;
  for (int l = 1; l < NLAYER; ++l) {
    gemm_xg<<<(unsigned)(MROWS / 16), 256, 0, stream>>>(
        cur, wp_ih + (size_t)(l - 1) * 65536, bsum + (size_t)l * GATES, xg16);
    lstm_scan<<<BATCH / 16, 256, 0, stream>>>(
        xg16, wp_hh + (size_t)l * 65536, nxt);
    _Float16* t = cur; cur = nxt; nxt = t;
  }

  // --- attention + FC ---------------------------------------------------
  attn_logits<<<(unsigned)(MROWS / 16), 256, 0, stream>>>(
      cur, wp_at, attn_b, v_w, v_b, logits);
  softmax_ctx_fc<<<BATCH, 256, 0, stream>>>(cur, logits, fc_w, fc_b, out);
}
